// NetTransModel_55362128445543
// MI455X (gfx1250) — matrix operand order repaired and checked by reading, not compiled, not run on hardware
//
#include <hip/hip_runtime.h>
#include <hip/hip_bf16.h>
#include <math.h>

#define N_NODES 15000
#define E_EDGES 180000
#define IN_F    256
#define HID     128
#define S_SUP   12750            // ceil(0.85 * 15000)
#define EA_EDGES (E_EDGES + S_SUP)
#define K_ITERS 10
#define ALPHA_V 0.5f

typedef __attribute__((ext_vector_type(16))) __bf16 v16bf;
typedef __attribute__((ext_vector_type(8)))  float  v8f;

// ---------------------------------------------------------------- utilities
__global__ void k_zero_f(float* p, int n) {
  int i = blockIdx.x * blockDim.x + threadIdx.x;
  if (i < n) p[i] = 0.0f;
}
__global__ void k_copy_f(float* dst, const float* src, int n) {
  int i = blockIdx.x * blockDim.x + threadIdx.x;
  if (i < n) dst[i] = src[i];
}
__global__ void k_relu_f(float* p, int n) {
  int i = blockIdx.x * blockDim.x + threadIdx.x;
  if (i < n) p[i] = fmaxf(p[i], 0.0f);
}

__device__ inline unsigned fenc(float f) {
  unsigned u = __float_as_uint(f);
  return (u & 0x80000000u) ? ~u : (u | 0x80000000u);
}
__device__ inline float fdec(unsigned u) {
  unsigned v = (u & 0x80000000u) ? (u & 0x7FFFFFFFu) : ~u;
  return __uint_as_float(v);
}
__global__ void k_fill_enc(unsigned* p, int n) {
  int i = blockIdx.x * blockDim.x + threadIdx.x;
  if (i < n) p[i] = fenc(-1e30f);
}

__device__ inline float rng_u01(unsigned i) {
  unsigned x = i + 0x9E3779B9u;
  x ^= x >> 16; x *= 0x7feb352du;
  x ^= x >> 15; x *= 0x846ca68bu;
  x ^= x >> 16;
  float u = (float)(x >> 8) * (1.0f / 16777216.0f);
  return fminf(fmaxf(u, 1e-6f), 1.0f - 1e-6f);
}

// -------------------------------------------- weight pre-pack to bf16 frags
// W[K][Nc] f32 row-major -> Wp: bf16, element ((kt*ntiles+nt)*32 + lane)*16 + j
// lane = khalf*16 + c ; fragment K-index = kt*32 + (j<8 ? khalf*8 + j : 16 + khalf*8 + j-8) ; col = nt*16+c.
// Requires K*Nc % 512 == 0 (holds: 256x128, 128x128).
__global__ void k_pack_w(const float* __restrict__ W, __bf16* __restrict__ Wp,
                         int K, int Nc) {
  int t = blockIdx.x * blockDim.x + threadIdx.x;
  if (t >= K * Nc) return;
  int j     = t & 15;
  int lane  = (t >> 4) & 31;
  int tile  = t >> 9;
  int ntls  = Nc >> 4;
  int nt    = tile % ntls;
  int kt    = tile / ntls;
  int k     = kt * 32 + (j < 8 ? (lane >> 4) * 8 + j : 16 + (lane >> 4) * 8 + (j - 8));
  int n     = nt * 16 + (lane & 15);
  Wp[t] = (__bf16)W[k * Nc + n];
}

// ------------------------------------------------ WMMA bf16 GEMM: C = A @ Wp
// A: [M,K] f32 row-major; Wp: pre-packed bf16 fragments; C: [M,Nc] f32.
// block = 128 threads = 4 waves stacked along M; each wave -> 16x64 strip
// (one A fragment reused by 4 B fragments -> 4 v_wmma per k-step).
__global__ void k_gemm_bf16(const float* __restrict__ A,
                            const __bf16* __restrict__ Wp,
                            float* __restrict__ C, int M, int K, int Nc) {
  int wave = threadIdx.x >> 5;
  int lane = threadIdx.x & 31;
  int m0 = (blockIdx.x * 4 + wave) * 16;
  int n0 = blockIdx.y * 64;
  int half = lane >> 4;
  int r    = lane & 15;
  int row  = m0 + r; if (row >= M) row = M - 1;   // clamp: rows >= M never stored
  int ntls = Nc >> 4;
  const v16bf* wp = (const v16bf*)Wp;
  v8f acc0 = {}, acc1 = {}, acc2 = {}, acc3 = {};
  for (int k0 = 0; k0 < K; k0 += 32) {
    // A fragment: two contiguous 8-float runs per lane -> 4x b128
    const float* ar = A + (size_t)row * K + k0 + half * 8;
    float4 a0 = *(const float4*)(ar);
    float4 a1 = *(const float4*)(ar + 4);
    float4 a2 = *(const float4*)(ar + 16);
    float4 a3 = *(const float4*)(ar + 20);
    v16bf av;
    av[0]=(__bf16)a0.x;  av[1]=(__bf16)a0.y;  av[2]=(__bf16)a0.z;  av[3]=(__bf16)a0.w;
    av[4]=(__bf16)a1.x;  av[5]=(__bf16)a1.y;  av[6]=(__bf16)a1.z;  av[7]=(__bf16)a1.w;
    av[8]=(__bf16)a2.x;  av[9]=(__bf16)a2.y;  av[10]=(__bf16)a2.z; av[11]=(__bf16)a2.w;
    av[12]=(__bf16)a3.x; av[13]=(__bf16)a3.y; av[14]=(__bf16)a3.z; av[15]=(__bf16)a3.w;
    // B fragments: contiguous 32B per lane, pre-packed
    size_t bidx = ((size_t)(k0 >> 5) * ntls + (n0 >> 4)) * 32 + lane;
    v16bf b0 = wp[bidx];
    v16bf b1 = wp[bidx + 32];
    v16bf b2 = wp[bidx + 64];
    v16bf b3 = wp[bidx + 96];
    acc0 = __builtin_amdgcn_wmma_f32_16x16x32_bf16(false, av, false, b0, (short)0, acc0, false, false);
    acc1 = __builtin_amdgcn_wmma_f32_16x16x32_bf16(false, av, false, b1, (short)0, acc1, false, false);
    acc2 = __builtin_amdgcn_wmma_f32_16x16x32_bf16(false, av, false, b2, (short)0, acc2, false, false);
    acc3 = __builtin_amdgcn_wmma_f32_16x16x32_bf16(false, av, false, b3, (short)0, acc3, false, false);
  }
#pragma unroll
  for (int i = 0; i < 8; ++i) {
    int rr = m0 + half * 8 + i;          // C layout: VGPR i -> M = i + 8*half
    if (rr < M) {
      float* cr = C + (size_t)rr * Nc + n0 + r;
      cr[0]  = acc0[i];
      cr[16] = acc1[i];
      cr[32] = acc2[i];
      cr[48] = acc3[i];
    }
  }
}

// ------------------------------------------------------------ sparse pieces
__global__ void k_seg_add(const int* idx, const float* w, float* acc, int n) {
  int e = blockIdx.x * blockDim.x + threadIdx.x;
  if (e < n) atomicAdd(&acc[idx[e]], w[e]);
}
__global__ void k_dinv(const float* deg, float* dinv, int n, float fill) {
  int i = blockIdx.x * blockDim.x + threadIdx.x;
  if (i >= n) return;
  float d = deg[i] + fill;
  dinv[i] = (d > 0.0f) ? rsqrtf(d) : 0.0f;
}
__global__ void k_edge_norm(const int* s, const int* d, const float* w,
                            const float* dinv, float* norm, int n) {
  int e = blockIdx.x * blockDim.x + threadIdx.x;
  if (e < n) norm[e] = dinv[s[e]] * w[e] * dinv[d[e]];
}
// Y[d[e], :] += coef[e] * X[s[e], :]  for D == HID == 128.
// thread = (edge, 4-feature chunk): one b128 gather + 4 atomic f32 adds.
__global__ void k_scatter_vec128(const int* __restrict__ s, const int* __restrict__ d,
                                 const float* __restrict__ coef,
                                 const float* __restrict__ X, float* __restrict__ Y,
                                 int n) {
  int t = blockIdx.x * blockDim.x + threadIdx.x;
  if (t >= n * 32) return;
  int e = t >> 5;
  int q = (t & 31) << 2;                 // feature offset 0..124 step 4
  float c = coef[e];
  if (c == 0.0f) return;
  const float4 v = *(const float4*)(X + ((size_t)s[e] << 7) + q);
  float* y = Y + ((size_t)d[e] << 7) + q;
  atomicAdd(y + 0, c * v.x);
  atomicAdd(y + 1, c * v.y);
  atomicAdd(y + 2, c * v.z);
  atomicAdd(y + 3, c * v.w);
}
// scalar-feature variant (D == 1), used by the score conv
__global__ void k_scatter_scalar(const int* s, const int* d, const float* coef,
                                 const float* X, float* Y, int n) {
  int e = blockIdx.x * blockDim.x + threadIdx.x;
  if (e >= n) return;
  float c = coef[e];
  if (c != 0.0f) atomicAdd(&Y[d[e]], c * X[s[e]]);
}
__global__ void k_combine(const float* znext, const float* h, float* z, int n) {
  int i = blockIdx.x * blockDim.x + threadIdx.x;
  if (i < n) z[i] = (1.0f - ALPHA_V) * znext[i] + ALPHA_V * h[i];
}
// out[i,:] += dinv[i]^2 * hX[i,:] + bias[:]   (D == 128, float4)
__global__ void k_selfbias128(float* __restrict__ out, const float* __restrict__ dinv,
                              const float* __restrict__ hX, const float* __restrict__ bias,
                              int n) {
  int t = blockIdx.x * blockDim.x + threadIdx.x;
  if (t >= n * 32) return;
  int i = t >> 5;
  int q = (t & 31) << 2;
  float dv = dinv[i];
  float s2 = dv * dv;
  size_t ofs = ((size_t)i << 7) + q;
  float4 o = *(const float4*)(out + ofs);
  float4 hx = *(const float4*)(hX + ofs);
  float4 b = *(const float4*)(bias + q);
  o.x += s2 * hx.x + b.x;
  o.y += s2 * hx.y + b.y;
  o.z += s2 * hx.z + b.z;
  o.w += s2 * hx.w + b.w;
  *(float4*)(out + ofs) = o;
}
__global__ void k_selfbias_scalar(float* out, const float* dinv, const float* hX,
                                  const float* bias, int n) {
  int i = blockIdx.x * blockDim.x + threadIdx.x;
  if (i >= n) return;
  float dv = dinv[i];
  out[i] += dv * dv * hX[i] + bias[0];
}
// masked edge weights
__global__ void k_effw(const int* s, const int* d, const float* ew,
                       const int* in_sup, float* out, int n, int mode) {
  int e = blockIdx.x * blockDim.x + threadIdx.x;
  if (e >= n) return;
  int ss = s[e], dd = d[e];
  bool m = (ss != dd);
  if (mode >= 2) m = m && (in_sup[dd] != 0);
  out[e] = m ? ew[e] : 0.0f;
}

// hs = xa @ score_W  (Nc = 1)
__global__ void k_gemv_score(const float* xa, const float* sw, float* hs, int n) {
  int i = blockIdx.x * blockDim.x + threadIdx.x;
  if (i >= n) return;
  float acc = 0.0f;
#pragma unroll 4
  for (int j = 0; j < HID; ++j) acc += xa[i * HID + j] * sw[j];
  hs[i] = acc;
}

// exact descending-stable rank (== position in argsort(-score))
__global__ void k_rank(const float* score, int* rank, int n) {
  __shared__ float sc[256];
  int i = blockIdx.x * 256 + threadIdx.x;
  float si = (i < n) ? score[i] : 0.0f;
  int rk = 0;
  for (int base = 0; base < n; base += 256) {
    int j = base + threadIdx.x;
    sc[threadIdx.x] = (j < n) ? score[j] : -1e38f;
    __syncthreads();
    int lim = min(256, n - base);
    if (i < n) {
      for (int jj = 0; jj < lim; ++jj) {
        float sj = sc[jj];
        rk += (sj > si) || ((sj == si) && (base + jj < i));
      }
    }
    __syncthreads();
  }
  if (i < n) rank[i] = rk;
}
__global__ void k_build_pos(const int* rank, int* in_sup, int* pos,
                            int* node_of_pos, int n) {
  int i = blockIdx.x * blockDim.x + threadIdx.x;
  if (i >= n) return;
  int rk = rank[i];
  int sup = (rk < S_SUP) ? 1 : 0;
  in_sup[i] = sup;
  pos[i] = sup ? rk : 0;
  if (sup) node_of_pos[rk] = i;
}
__global__ void k_sup_gather(const float* temp_x, const int* node_of_pos,
                             float* sup_x, int n) {   // n = S*32 (float4 chunks)
  int t = blockIdx.x * blockDim.x + threadIdx.x;
  if (t >= n) return;
  int p = t >> 5, q = (t & 31) << 2;
  float4 v = *(const float4*)(temp_x + ((size_t)node_of_pos[p] << 7) + q);
  v.x = fmaxf(v.x, 0.0f); v.y = fmaxf(v.y, 0.0f);
  v.z = fmaxf(v.z, 0.0f); v.w = fmaxf(v.w, 0.0f);
  *(float4*)(sup_x + ((size_t)p << 7) + q) = v;
}

// candidate assignment edges (E graph edges + S self edges)
__global__ void k_build_assign(const int* s, const int* d, const float* ew,
                               const int* in_sup, const int* pos, const int* node_of_pos,
                               int* a_src, int* a_dst, int* a_mask, float* a_w, int n) {
  int e = blockIdx.x * blockDim.x + threadIdx.x;
  if (e >= n) return;
  if (e < E_EDGES) {
    int ss = s[e], dd = d[e];
    bool m1 = (ss != dd) && (in_sup[dd] != 0);
    a_src[e] = ss;
    a_dst[e] = m1 ? pos[dd] : 0;
    a_mask[e] = m1 ? 1 : 0;
    a_w[e] = m1 ? ew[e] : 0.0f;
  } else {
    int p = e - E_EDGES;
    a_src[e] = node_of_pos[p];
    a_dst[e] = p;
    a_mask[e] = 1;
    a_w[e] = 1.0f;
  }
}
// aw = sigmoid(<temp_x[a_src], sup_x[a_dst]>) * a_w   (wave per edge, float4)
__global__ void k_assign_dot(const int* a_src, const int* a_dst, const float* a_w,
                             const float* tx, const float* sx, float* aw, int n) {
  int e = blockIdx.x * 8 + (threadIdx.x >> 5);
  if (e >= n) return;
  int lane = threadIdx.x & 31;
  const float4* pa = (const float4*)(tx + ((long long)a_src[e] << 7)) + lane;
  const float4* pb = (const float4*)(sx + ((long long)a_dst[e] << 7)) + lane;
  float4 va = *pa, vb = *pb;                 // 32 lanes x 4 floats = 128
  float sum = va.x * vb.x + va.y * vb.y + va.z * vb.z + va.w * vb.w;
#pragma unroll
  for (int off = 16; off > 0; off >>= 1) sum += __shfl_down(sum, off);
  if (lane == 0) aw[e] = (1.0f / (1.0f + expf(-sum))) * a_w[e];
}
__global__ void k_gumbel(const float* aw, const float* rowsum, const int* a_src,
                         const int* a_mask, const float* temperature,
                         float* logits, unsigned* menc, int n) {
  int e = blockIdx.x * blockDim.x + threadIdx.x;
  if (e >= n) return;
  int s = a_src[e];
  float v = aw[e] / fmaxf(rowsum[s], 1e-12f);
  float u = rng_u01((unsigned)e);
  float g = -logf(-logf(u));
  float t = temperature[0];
  float lg = a_mask[e] ? (logf(v + 1e-20f) + g) / t : -1e9f;
  logits[e] = lg;
  atomicMax(&menc[s], fenc(lg));
}
__global__ void k_expsum(const float* logits, const unsigned* menc, const int* a_src,
                         const int* a_mask, float* eb, float* denom, int n) {
  int e = blockIdx.x * blockDim.x + threadIdx.x;
  if (e >= n) return;
  int s = a_src[e];
  float m = fdec(menc[s]);
  m = (m > -1e8f) ? m : 0.0f;
  float ev = a_mask[e] ? expf(logits[e] - m) : 0.0f;
  eb[e] = ev;
  atomicAdd(&denom[s], ev);
}
__global__ void k_aw_final(const float* eb, const float* denom, const int* a_src,
                           const int* a_mask, float* aw2, int n) {
  int e = blockIdx.x * blockDim.x + threadIdx.x;
  if (e >= n) return;
  float v = eb[e] / fmaxf(denom[a_src[e]], 1e-12f);
  const float thresh = 0.5f / (float)S_SUP;
  aw2[e] = (a_mask[e] && v >= thresh) ? v : 0.0f;
}
__global__ void k_xnew(const float* sup_x, const float* x_pool, float* x_new, int n) {
  int i = blockIdx.x * blockDim.x + threadIdx.x;
  if (i < n) x_new[i] = 0.5f * (sup_x[i] + x_pool[i]);
}
// pooled-graph edges (indices already remapped to supernode ids, 0 when masked)
__global__ void k_out_edges(const int* s, const int* d, const float* ew,
                            const int* in_sup, const int* pos,
                            float* wo, int* osrc, int* odst, int n) {
  int e = blockIdx.x * blockDim.x + threadIdx.x;
  if (e >= n) return;
  int ss = s[e], dd = d[e];
  bool m = (ss != dd) && (in_sup[dd] != 0) && (in_sup[ss] != 0);
  int ps = m ? pos[ss] : 0, pd = m ? pos[dd] : 0;
  m = m && (ps != pd);
  wo[e] = m ? ew[e] : 0.0f;
  osrc[e] = m ? ps : 0;
  odst[e] = m ? pd : 0;
}

// ------------------------------------------------------------------ driver
static inline size_t alignup(size_t x) { return (x + 255) & ~(size_t)255; }
static inline int cdiv(long long a, int b) { return (int)((a + b - 1) / b); }

extern "C" void kernel_launch(void* const* d_in, const int* in_sizes, int n_in,
                              void* d_out, int out_size, void* d_ws, size_t ws_size,
                              hipStream_t stream) {
  const float* x        = (const float*)d_in[0];
  const int*   ei       = (const int*)d_in[1];
  const float* ew       = (const float*)d_in[2];
  const float* temp_p   = (const float*)d_in[7];
  const float* W0       = (const float*)d_in[8];
  const float* score_W  = (const float*)d_in[9];
  const float* score_b  = (const float*)d_in[10];
  const float* aggr_W   = (const float*)d_in[11];
  const float* aggr_b   = (const float*)d_in[12];
  const float* down1_W  = (const float*)d_in[13];
  const float* down1_b  = (const float*)d_in[14];
  float* out = (float*)d_out;

  const int* src = ei;             // edge_index[0]
  const int* dst = ei + E_EDGES;   // edge_index[1]

  // ---- workspace bump allocator (deterministic)
  char* base = (char*)d_ws;
  size_t off = 0;
  auto alloc = [&](size_t bytes) -> void* {
    void* p = base + off;
    off = alignup(off + bytes);
    return p;
  };
  const size_t NH = (size_t)N_NODES * HID, SH = (size_t)S_SUP * HID;
  float* h     = (float*)alloc(NH * 4);
  float* z     = (float*)alloc(NH * 4);
  float* znext = (float*)alloc(NH * 4);
  float* hA    = (float*)alloc(NH * 4);
  float* tempx = (float*)alloc(NH * 4);
  float* supx  = (float*)alloc(SH * 4);
  float* xpool = (float*)alloc(SH * 4);
  float* xnew  = (float*)alloc(SH * 4);
  float* hN    = (float*)alloc(SH * 4);
  float* deg   = (float*)alloc((size_t)N_NODES * 4);
  float* dinv0 = (float*)alloc((size_t)N_NODES * 4);
  float* dinvS = (float*)alloc((size_t)N_NODES * 4);
  float* dinv1 = (float*)alloc((size_t)N_NODES * 4);
  float* dinvO = (float*)alloc((size_t)S_SUP * 4);
  float* norm0 = (float*)alloc((size_t)E_EDGES * 4);
  float* normS = (float*)alloc((size_t)E_EDGES * 4);
  float* norm1 = (float*)alloc((size_t)E_EDGES * 4);
  float* normO = (float*)alloc((size_t)E_EDGES * 4);
  float* ewEff = (float*)alloc((size_t)E_EDGES * 4);
  float* wo    = (float*)alloc((size_t)E_EDGES * 4);
  float* hs    = (float*)alloc((size_t)N_NODES * 4);
  float* score = (float*)alloc((size_t)N_NODES * 4);
  float* rowsum= (float*)alloc((size_t)N_NODES * 4);
  float* denom = (float*)alloc((size_t)N_NODES * 4);
  float* aw    = (float*)alloc((size_t)EA_EDGES * 4);
  float* a_w   = (float*)alloc((size_t)EA_EDGES * 4);
  float* logits= (float*)alloc((size_t)EA_EDGES * 4);
  float* eb    = (float*)alloc((size_t)EA_EDGES * 4);
  float* aw2   = (float*)alloc((size_t)EA_EDGES * 4);
  int*   rank  = (int*)alloc((size_t)N_NODES * 4);
  int*   insup = (int*)alloc((size_t)N_NODES * 4);
  int*   pos   = (int*)alloc((size_t)N_NODES * 4);
  int*   nodep = (int*)alloc((size_t)S_SUP * 4);
  int*   a_src = (int*)alloc((size_t)EA_EDGES * 4);
  int*   a_dst = (int*)alloc((size_t)EA_EDGES * 4);
  int*   a_msk = (int*)alloc((size_t)EA_EDGES * 4);
  int*   osrc  = (int*)alloc((size_t)E_EDGES * 4);
  int*   odst  = (int*)alloc((size_t)E_EDGES * 4);
  unsigned* menc = (unsigned*)alloc((size_t)N_NODES * 4);
  __bf16* Wp0 = (__bf16*)alloc((size_t)IN_F * HID * 2);
  __bf16* WpA = (__bf16*)alloc((size_t)HID * HID * 2);
  __bf16* WpD = (__bf16*)alloc((size_t)HID * HID * 2);
  (void)ws_size; (void)n_in; (void)in_sizes; (void)out_size;

  const int T = 256;
  #define G(n) cdiv((long long)(n), T), T, 0, stream

  // ---- 0) pre-pack the three weight matrices into WMMA fragment layout
  hipLaunchKernelGGL(k_pack_w, G(IN_F * HID), W0, Wp0, IN_F, HID);
  hipLaunchKernelGGL(k_pack_w, G(HID * HID), aggr_W, WpA, HID, HID);
  hipLaunchKernelGGL(k_pack_w, G(HID * HID), down1_W, WpD, HID, HID);

  // ---- 1) h = x @ W0  (WMMA bf16)
  hipLaunchKernelGGL(k_gemm_bf16, dim3(cdiv(N_NODES, 64), HID / 64), dim3(128), 0, stream,
                     x, Wp0, h, N_NODES, IN_F, HID);

  // ---- 2) APPNP normalization (no self loops)
  hipLaunchKernelGGL(k_zero_f, G(N_NODES), deg, N_NODES);
  hipLaunchKernelGGL(k_seg_add, G(E_EDGES), dst, ew, deg, E_EDGES);
  hipLaunchKernelGGL(k_dinv, G(N_NODES), deg, dinv0, N_NODES, 0.0f);
  hipLaunchKernelGGL(k_edge_norm, G(E_EDGES), src, dst, ew, dinv0, norm0, E_EDGES);

  // ---- 3) APPNP iterations: z = 0.5*prop(z) + 0.5*h
  hipLaunchKernelGGL(k_copy_f, G(NH), z, h, (int)NH);
  for (int it = 0; it < K_ITERS; ++it) {
    hipLaunchKernelGGL(k_zero_f, G(NH), znext, (int)NH);
    hipLaunchKernelGGL(k_scatter_vec128, G(E_EDGES * 32),
                       src, dst, norm0, z, znext, E_EDGES);
    hipLaunchKernelGGL(k_combine, G(NH), znext, h, z, (int)NH);
  }
  hipLaunchKernelGGL(k_relu_f, G(NH), z, (int)NH);   // z := xa

  // ---- 4) score GCNConv (mask self loops, fill=1)
  hipLaunchKernelGGL(k_effw, G(E_EDGES), src, dst, ew, insup, ewEff, E_EDGES, 1);
  hipLaunchKernelGGL(k_zero_f, G(N_NODES), deg, N_NODES);
  hipLaunchKernelGGL(k_seg_add, G(E_EDGES), dst, ewEff, deg, E_EDGES);
  hipLaunchKernelGGL(k_dinv, G(N_NODES), deg, dinvS, N_NODES, 1.0f);
  hipLaunchKernelGGL(k_edge_norm, G(E_EDGES), src, dst, ewEff, dinvS, normS, E_EDGES);
  hipLaunchKernelGGL(k_gemv_score, G(N_NODES), z, score_W, hs, N_NODES);
  hipLaunchKernelGGL(k_zero_f, G(N_NODES), score, N_NODES);
  hipLaunchKernelGGL(k_scatter_scalar, G(E_EDGES), src, dst, normS, hs, score, E_EDGES);
  hipLaunchKernelGGL(k_selfbias_scalar, G(N_NODES), score, dinvS, hs, score_b, N_NODES);

  // ---- 5) exact top-k via O(N^2) ranking
  hipLaunchKernelGGL(k_rank, dim3(cdiv(N_NODES, 256)), dim3(256), 0, stream,
                     score, rank, N_NODES);
  hipLaunchKernelGGL(k_build_pos, G(N_NODES), rank, insup, pos, nodep, N_NODES);

  // ---- 6) aggr GCNConv (filter_target) -> temp_x, sup_x
  hipLaunchKernelGGL(k_effw, G(E_EDGES), src, dst, ew, insup, ewEff, E_EDGES, 2);
  hipLaunchKernelGGL(k_zero_f, G(N_NODES), deg, N_NODES);
  hipLaunchKernelGGL(k_seg_add, G(E_EDGES), dst, ewEff, deg, E_EDGES);
  hipLaunchKernelGGL(k_dinv, G(N_NODES), deg, dinv1, N_NODES, 1.0f);
  hipLaunchKernelGGL(k_edge_norm, G(E_EDGES), src, dst, ewEff, dinv1, norm1, E_EDGES);
  hipLaunchKernelGGL(k_gemm_bf16, dim3(cdiv(N_NODES, 64), HID / 64), dim3(128), 0, stream,
                     z, WpA, hA, N_NODES, HID, HID);
  hipLaunchKernelGGL(k_zero_f, G(NH), tempx, (int)NH);
  hipLaunchKernelGGL(k_scatter_vec128, G(E_EDGES * 32),
                     src, dst, norm1, hA, tempx, E_EDGES);
  hipLaunchKernelGGL(k_selfbias128, G(N_NODES * 32), tempx, dinv1, hA, aggr_b, N_NODES);
  hipLaunchKernelGGL(k_sup_gather, G(S_SUP * 32), tempx, nodep, supx, S_SUP * 32);

  // ---- 7) soft assignment + gumbel softmax + pooling
  hipLaunchKernelGGL(k_build_assign, G(EA_EDGES), src, dst, ew, insup, pos, nodep,
                     a_src, a_dst, a_msk, a_w, EA_EDGES);
  hipLaunchKernelGGL(k_assign_dot, dim3(cdiv(EA_EDGES, 8)), dim3(256), 0, stream,
                     a_src, a_dst, a_w, tempx, supx, aw, EA_EDGES);
  hipLaunchKernelGGL(k_zero_f, G(N_NODES), rowsum, N_NODES);
  hipLaunchKernelGGL(k_seg_add, G(EA_EDGES), a_src, aw, rowsum, EA_EDGES);
  hipLaunchKernelGGL(k_fill_enc, G(N_NODES), menc, N_NODES);
  hipLaunchKernelGGL(k_gumbel, G(EA_EDGES), aw, rowsum, a_src, a_msk, temp_p,
                     logits, menc, EA_EDGES);
  hipLaunchKernelGGL(k_zero_f, G(N_NODES), denom, N_NODES);
  hipLaunchKernelGGL(k_expsum, G(EA_EDGES), logits, menc, a_src, a_msk, eb, denom, EA_EDGES);
  hipLaunchKernelGGL(k_aw_final, G(EA_EDGES), eb, denom, a_src, a_msk, aw2, EA_EDGES);
  hipLaunchKernelGGL(k_zero_f, G(SH), xpool, (int)SH);
  hipLaunchKernelGGL(k_scatter_vec128, G(EA_EDGES * 32),
                     a_src, a_dst, aw2, tempx, xpool, EA_EDGES);
  hipLaunchKernelGGL(k_xnew, G(SH), supx, xpool, xnew, (int)SH);

  // ---- 8) final GCNConv on pooled supergraph + relu
  hipLaunchKernelGGL(k_out_edges, G(E_EDGES), src, dst, ew, insup, pos,
                     wo, osrc, odst, E_EDGES);
  hipLaunchKernelGGL(k_zero_f, G(S_SUP), deg, S_SUP);
  hipLaunchKernelGGL(k_seg_add, G(E_EDGES), odst, wo, deg, E_EDGES);
  hipLaunchKernelGGL(k_dinv, G(S_SUP), deg, dinvO, S_SUP, 1.0f);
  hipLaunchKernelGGL(k_edge_norm, G(E_EDGES), osrc, odst, wo, dinvO, normO, E_EDGES);
  hipLaunchKernelGGL(k_gemm_bf16, dim3(cdiv(S_SUP, 64), HID / 64), dim3(128), 0, stream,
                     xnew, WpD, hN, S_SUP, HID, HID);
  hipLaunchKernelGGL(k_zero_f, G(SH), out, (int)SH);
  hipLaunchKernelGGL(k_scatter_vec128, G(E_EDGES * 32),
                     osrc, odst, normO, hN, out, E_EDGES);
  hipLaunchKernelGGL(k_selfbias128, G(S_SUP * 32), out, dinvO, hN, down1_b, S_SUP);
  hipLaunchKernelGGL(k_relu_f, G(SH), out, (int)SH);
  #undef G
}
